// ReconfiguredAttention_33535104647238
// MI455X (gfx1250) — matrix operand order repaired and checked by reading, not compiled, not run on hardware
//
#include <hip/hip_runtime.h>
#include <hip/hip_fp16.h>

typedef __attribute__((ext_vector_type(16))) _Float16 v16h;
typedef __attribute__((ext_vector_type(4)))  _Float16 v4h;
typedef __attribute__((ext_vector_type(8)))  float    v8f;

#define C_DIM      256
#define NHEAD      8
#define HD         32
#define L11        4096
#define L12        1024
#define L2Q        1024
#define TOK_PER_B  6144      // L11 + L12 + L2Q
#define OUT_PER_B  5120      // L11 + L2Q
#define QKV_N      768
#define NBATCH     8
#define MTOT       (NBATCH * TOK_PER_B)   // 49152
#define OUT_MTOT   (NBATCH * OUT_PER_B)   // 40960

// LDS row strides (halves). 40 halves = 80B (16B-aligned rows, odd dword phase
// per 16-row fragment -> conflict-free b128 reads). 72 halves = 144B likewise.
#define LDA   40    // 32-wide f16 tiles (Q, K, GEMM A, GEMM B-transposed)
#define LDP   72    // 64-wide f16 tiles (P, V-transposed)
#define LDS_S 65    // 64-wide f32 score rows (odd dword stride: conflict-free)

// ---------------- WMMA helpers (wave32, A/B use contiguous k = 16*(lane/16) + s) ----------------

static __device__ __forceinline__ v8f wmma16(v16h a, v16h b, v8f c) {
  return __builtin_amdgcn_wmma_f32_16x16x32_f16(
      false, a, false, b, (short)0, c, false, false);
}

// A fragment: 16x32 f16 tile from row-major LDS [rows][ld].
// lane 0..15: M=lane, K=0..15 ; lane 16..31: M=lane-16, K=16..31
static __device__ __forceinline__ v16h ldsA(const _Float16* s, int ld,
                                            int mBase, int kBase, int lane) {
  int m  = mBase + (lane & 15);
  int kh = (lane >> 4) * 16;
  v16h a;
#pragma unroll
  for (int i = 0; i < 8; ++i) a[i]     = s[m * ld + kBase + kh + i];
#pragma unroll
  for (int i = 0; i < 8; ++i) a[8 + i] = s[m * ld + kBase + 8 + kh + i];
  return a;
}

// B fragment (32x16, KxN) from an N-major ("transposed") LDS tile [N][ld]:
// B(k,n) = s[n][k]  -> each lane reads 16 contiguous halves (2x b128).
static __device__ __forceinline__ v16h ldsBT(const _Float16* s, int ld,
                                             int kBase, int nBase, int lane) {
  int n  = nBase + (lane & 15);
  int k0 = kBase + (lane >> 4) * 16;
  v16h b;
#pragma unroll
  for (int i = 0; i < 16; ++i) b[i] = s[n * ld + k0 + i];
  return b;
}

// ---------------- Kernel 1: joint QKV projection ----------------

static __device__ __forceinline__ const float* rowPtrX(const float* xs1,
                                                       const float* xs2,
                                                       const float* xq, int row) {
  int b = row / TOK_PER_B;
  int t = row - b * TOK_PER_B;
  if (t < L11) return xs1 + ((size_t)b * L11 + t) * C_DIM;
  t -= L11;
  if (t < L12) return xs2 + ((size_t)b * L12 + t) * C_DIM;
  t -= L12;
  return xq + ((size_t)b * L2Q + t) * C_DIM;
}

__global__ __launch_bounds__(256) void qkv_gemm_kernel(
    const float* __restrict__ xs1, const float* __restrict__ xs2,
    const float* __restrict__ xq,  const float* __restrict__ W,
    const float* __restrict__ bias, _Float16* __restrict__ qkv) {
  __shared__ _Float16 sA[64 * LDA];     // 64 tokens x 32 K
  __shared__ _Float16 sBt[128 * LDA];   // 128 N x 32 K (transposed W tile)
  const int tid  = threadIdx.x;
  const int lane = tid & 31;
  const int wave = tid >> 5;
  const int wm = wave & 3, wn = wave >> 2;          // 4 (M) x 2 (N) waves
  const int mBase = blockIdx.x * 64;
  const int nBase = blockIdx.y * 128;

  v8f acc[4] = {};
  for (int kb = 0; kb < C_DIM; kb += 32) {
    // A: 64 rows x 32 K, float4 loads -> v4h stores (2 chunks / thread)
#pragma unroll
    for (int j = 0; j < 2; ++j) {
      int cid = tid + j * 256;              // 0..511
      int r = cid >> 3, c = cid & 7;
      float4 f = *(const float4*)(rowPtrX(xs1, xs2, xq, mBase + r) + kb + c * 4);
      v4h hv; hv[0] = (_Float16)f.x; hv[1] = (_Float16)f.y;
              hv[2] = (_Float16)f.z; hv[3] = (_Float16)f.w;
      *(v4h*)(sA + r * LDA + c * 4) = hv;
    }
    // B (transposed): float4 loads of W row, scatter 4 halves into [n][k]
#pragma unroll
    for (int j = 0; j < 4; ++j) {
      int cid = tid + j * 256;              // 0..1023
      int k = cid >> 5, nc = cid & 31;
      float4 f = *(const float4*)(W + (size_t)(kb + k) * QKV_N + nBase + nc * 4);
      sBt[(nc * 4 + 0) * LDA + k] = (_Float16)f.x;
      sBt[(nc * 4 + 1) * LDA + k] = (_Float16)f.y;
      sBt[(nc * 4 + 2) * LDA + k] = (_Float16)f.z;
      sBt[(nc * 4 + 3) * LDA + k] = (_Float16)f.w;
    }
    if (kb + 32 < C_DIM)   // prefetch next W K-slice (global_prefetch_b8)
      __builtin_prefetch(W + (size_t)(kb + 32 + (tid >> 5)) * QKV_N + nBase +
                             (tid & 31) * 4, 0, 1);
    __syncthreads();
    v16h a = ldsA(sA, LDA, wm * 16, 0, lane);
#pragma unroll
    for (int ni = 0; ni < 4; ++ni) {
      v16h b = ldsBT(sBt, LDA, 0, wn * 64 + ni * 16, lane);
      acc[ni] = wmma16(a, b, acc[ni]);
    }
    __syncthreads();
  }
  const int mOff = (lane >> 4) * 8;
  const int n0   = lane & 15;
#pragma unroll
  for (int ni = 0; ni < 4; ++ni)
#pragma unroll
    for (int r = 0; r < 8; ++r) {
      int row = mBase + wm * 16 + mOff + r;
      int col = nBase + wn * 64 + ni * 16 + n0;
      qkv[(size_t)row * QKV_N + col] = (_Float16)(acc[ni][r] + bias[col]);
    }
}

// ---------------- Kernel 2: 8x8 windowed self-attention on xs1 ----------------

__global__ __launch_bounds__(32) void win_attn_kernel(
    const _Float16* __restrict__ qkv, const float* __restrict__ relTable,
    _Float16* __restrict__ attn) {
  __shared__ _Float16 sQ[64 * LDA], sK[64 * LDA];
  __shared__ _Float16 sVt[32 * LDP];          // [channel][kvRow]
  __shared__ float    sS[64 * LDS_S];
  __shared__ _Float16 sP[64 * LDP];
  const int lane  = threadIdx.x;
  const int h     = blockIdx.x & 7;
  const int wid   = blockIdx.x >> 3;          // 512 windows
  const int batch = wid >> 6;
  const int win   = wid & 63;
  const int wr = win >> 3, wc = win & 7;

  // Vectorized load: 4 rows / iter, 8 lanes per row, v4h (8B) chunks
#pragma unroll 4
  for (int it = 0; it < 16; ++it) {
    int r  = it * 4 + (lane >> 3);
    int c4 = lane & 7;
    int lr = (wr * 8 + (r >> 3)) * 64 + (wc * 8 + (r & 7));   // image token
    const _Float16* base =
        qkv + ((size_t)batch * TOK_PER_B + lr) * QKV_N + h * HD;
    v4h qv = *(const v4h*)(base + c4 * 4);
    v4h kv = *(const v4h*)(base + 256 + c4 * 4);
    v4h vv = *(const v4h*)(base + 512 + c4 * 4);
    *(v4h*)(sQ + r * LDA + c4 * 4) = qv;
    *(v4h*)(sK + r * LDA + c4 * 4) = kv;
#pragma unroll
    for (int i = 0; i < 4; ++i) sVt[(c4 * 4 + i) * LDP + r] = vv[i];
  }
  __syncthreads();

  // S = Q * K^T : 4x4 tiles, K=32 single WMMA step
  const int mOff = (lane >> 4) * 8;
  const int n0   = lane & 15;
#pragma unroll
  for (int mi = 0; mi < 4; ++mi) {
    v16h a = ldsA(sQ, LDA, mi * 16, 0, lane);
#pragma unroll
    for (int ni = 0; ni < 4; ++ni) {
      v16h b = ldsBT(sK, LDA, 0, ni * 16, lane);
      v8f c = {};
      c = wmma16(a, b, c);
#pragma unroll
      for (int r2 = 0; r2 < 8; ++r2)
        sS[(mi * 16 + mOff + r2) * LDS_S + ni * 16 + n0] = c[r2];
    }
  }
  __syncthreads();

  // softmax with rel-pos bias: logit = s/hd + bias/sqrt(hd)  (2 rows / lane)
  const float inv  = 0.17677669529663687f;   // 1/sqrt(32)
  const float inv2 = inv * inv;              // 1/32
  for (int rr = 0; rr < 2; ++rr) {
    int r = lane + rr * 32;
    int qi = r >> 3, qj = r & 7;
    float mx = -1e30f;
    for (int k = 0; k < 64; ++k) {
      int ki = k >> 3, kj = k & 7;
      int ridx = (qi - ki + 7) * 15 + (qj - kj + 7);
      float lg = sS[r * LDS_S + k] * inv2 + relTable[ridx * NHEAD + h] * inv;
      sS[r * LDS_S + k] = lg;
      mx = fmaxf(mx, lg);
    }
    float sum = 0.f;
    for (int k = 0; k < 64; ++k) {
      float e = __expf(sS[r * LDS_S + k] - mx);
      sS[r * LDS_S + k] = e;
      sum += e;
    }
    float rs = 1.f / sum;
    for (int k = 0; k < 64; ++k)
      sP[r * LDP + k] = (_Float16)(sS[r * LDS_S + k] * rs);
  }
  __syncthreads();

  // O = P * V : 4x2 tiles, K=64 in two WMMA steps
#pragma unroll
  for (int mi = 0; mi < 4; ++mi)
#pragma unroll
    for (int ni = 0; ni < 2; ++ni) {
      v8f o = {};
#pragma unroll
      for (int kk = 0; kk < 2; ++kk) {
        v16h a = ldsA(sP, LDP, mi * 16, kk * 32, lane);
        v16h b = ldsBT(sVt, LDP, kk * 32, ni * 16, lane);
        o = wmma16(a, b, o);
      }
#pragma unroll
      for (int r2 = 0; r2 < 8; ++r2) {
        int row = mi * 16 + mOff + r2;
        int lr = (wr * 8 + (row >> 3)) * 64 + (wc * 8 + (row & 7));
        attn[((size_t)batch * OUT_PER_B + lr) * C_DIM + h * HD + ni * 16 + n0] =
            (_Float16)o[r2];
      }
    }
}

// ---------------- Kernel 3: cross attention (flash-style streaming) ----------------

__global__ __launch_bounds__(32) void cross_attn_kernel(
    const _Float16* __restrict__ qkv, _Float16* __restrict__ attn) {
  __shared__ _Float16 sQ[64 * LDA], sK[64 * LDA];
  __shared__ _Float16 sVt[32 * LDP];
  __shared__ float    sS[64 * LDS_S];
  __shared__ _Float16 sP[64 * LDP];
  __shared__ float    sM[64], sL[64], sAl[64];
  const int lane  = threadIdx.x;
  const int qc    = blockIdx.x & 15;          // 16 query chunks of 64
  const int h     = (blockIdx.x >> 4) & 7;
  const int batch = blockIdx.x >> 7;

  const int qTok = L11 + L12 + qc * 64;
#pragma unroll 4
  for (int it = 0; it < 16; ++it) {
    int r  = it * 4 + (lane >> 3);
    int c4 = lane & 7;
    const _Float16* base =
        qkv + ((size_t)batch * TOK_PER_B + qTok + r) * QKV_N + h * HD;
    *(v4h*)(sQ + r * LDA + c4 * 4) = *(const v4h*)(base + c4 * 4);
  }
  sM[lane] = -1e30f; sM[lane + 32] = -1e30f;
  sL[lane] = 0.f;    sL[lane + 32] = 0.f;
  __syncthreads();

  v16h aq[4];
#pragma unroll
  for (int mi = 0; mi < 4; ++mi) aq[mi] = ldsA(sQ, LDA, mi * 16, 0, lane);

  v8f o[4][2] = {};
  const float inv = 0.17677669529663687f;     // 1/sqrt(32)
  const int mOff = (lane >> 4) * 8;
  const int n0   = lane & 15;

  for (int t = 0; t < 32; ++t) {              // 2048 keys in tiles of 64
#pragma unroll 4
    for (int it = 0; it < 16; ++it) {
      int r  = it * 4 + (lane >> 3);
      int c4 = lane & 7;
      int j  = t * 64 + r;
      int tok = (j < L12) ? (L11 + j) : (L11 + L12 + (j - L12));
      const _Float16* base =
          qkv + ((size_t)batch * TOK_PER_B + tok) * QKV_N + h * HD;
      v4h kv = *(const v4h*)(base + 256 + c4 * 4);
      v4h vv = *(const v4h*)(base + 512 + c4 * 4);
      *(v4h*)(sK + r * LDA + c4 * 4) = kv;
#pragma unroll
      for (int i = 0; i < 4; ++i) sVt[(c4 * 4 + i) * LDP + r] = vv[i];
    }
    __syncthreads();

#pragma unroll
    for (int mi = 0; mi < 4; ++mi)
#pragma unroll
      for (int ni = 0; ni < 4; ++ni) {
        v16h b = ldsBT(sK, LDA, 0, ni * 16, lane);
        v8f c = {};
        c = wmma16(aq[mi], b, c);
#pragma unroll
        for (int r2 = 0; r2 < 8; ++r2)
          sS[(mi * 16 + mOff + r2) * LDS_S + ni * 16 + n0] = c[r2];
      }
    __syncthreads();

    // online softmax update (2 rows / lane)
    for (int rr = 0; rr < 2; ++rr) {
      int r = lane + rr * 32;
      float tm = -1e30f;
      for (int k = 0; k < 64; ++k) tm = fmaxf(tm, sS[r * LDS_S + k] * inv);
      float mold = sM[r];
      float mnew = fmaxf(mold, tm);
      float alpha = __expf(mold - mnew);
      float sum = 0.f;
      for (int k = 0; k < 64; ++k) {
        float e = __expf(sS[r * LDS_S + k] * inv - mnew);
        sP[r * LDP + k] = (_Float16)e;
        sum += e;
      }
      sM[r]  = mnew;
      sL[r]  = sL[r] * alpha + sum;
      sAl[r] = alpha;
    }
    __syncthreads();

    // rescale accumulators, then O += P * V
#pragma unroll
    for (int mi = 0; mi < 4; ++mi)
#pragma unroll
      for (int r2 = 0; r2 < 8; ++r2) {
        float al = sAl[mi * 16 + mOff + r2];
        o[mi][0][r2] *= al;
        o[mi][1][r2] *= al;
      }
#pragma unroll
    for (int mi = 0; mi < 4; ++mi)
#pragma unroll
      for (int kk = 0; kk < 2; ++kk) {
        v16h a = ldsA(sP, LDP, mi * 16, kk * 32, lane);
#pragma unroll
        for (int ni = 0; ni < 2; ++ni) {
          v16h b = ldsBT(sVt, LDP, kk * 32, ni * 16, lane);
          o[mi][ni] = wmma16(a, b, o[mi][ni]);
        }
      }
    __syncthreads();
  }

  // epilogue: divide by running sum, write rows [L11 .. L11+1024)
#pragma unroll
  for (int mi = 0; mi < 4; ++mi)
#pragma unroll
    for (int r2 = 0; r2 < 8; ++r2) {
      int row = mi * 16 + mOff + r2;
      float rl = 1.f / sL[row];
#pragma unroll
      for (int ni = 0; ni < 2; ++ni)
        attn[((size_t)batch * OUT_PER_B + L11 + qc * 64 + row) * C_DIM +
             h * HD + ni * 16 + n0] = (_Float16)(o[mi][ni][r2] * rl);
    }
}

// ---------------- Kernel 4: output projection ----------------

__global__ __launch_bounds__(256) void proj_gemm_kernel(
    const _Float16* __restrict__ attn, const float* __restrict__ W,
    const float* __restrict__ bias, float* __restrict__ out) {
  __shared__ _Float16 sA[64 * LDA];
  __shared__ _Float16 sBt[128 * LDA];
  const int tid  = threadIdx.x;
  const int lane = tid & 31;
  const int wave = tid >> 5;
  const int wm = wave & 3, wn = wave >> 2;
  const int mBase = blockIdx.x * 64;
  const int nBase = blockIdx.y * 128;

  v8f acc[4] = {};
  for (int kb = 0; kb < C_DIM; kb += 32) {
#pragma unroll
    for (int j = 0; j < 2; ++j) {
      int cid = tid + j * 256;
      int r = cid >> 3, c = cid & 7;
      *(v4h*)(sA + r * LDA + c * 4) =
          *(const v4h*)(attn + (size_t)(mBase + r) * C_DIM + kb + c * 4);
    }
#pragma unroll
    for (int j = 0; j < 4; ++j) {
      int cid = tid + j * 256;
      int k = cid >> 5, nc = cid & 31;
      float4 f = *(const float4*)(W + (size_t)(kb + k) * C_DIM + nBase + nc * 4);
      sBt[(nc * 4 + 0) * LDA + k] = (_Float16)f.x;
      sBt[(nc * 4 + 1) * LDA + k] = (_Float16)f.y;
      sBt[(nc * 4 + 2) * LDA + k] = (_Float16)f.z;
      sBt[(nc * 4 + 3) * LDA + k] = (_Float16)f.w;
    }
    if (kb + 32 < C_DIM)
      __builtin_prefetch(W + (size_t)(kb + 32 + (tid >> 5)) * C_DIM + nBase +
                             (tid & 31) * 4, 0, 1);
    __syncthreads();
    v16h a = ldsA(sA, LDA, wm * 16, 0, lane);
#pragma unroll
    for (int ni = 0; ni < 4; ++ni) {
      v16h b = ldsBT(sBt, LDA, 0, wn * 64 + ni * 16, lane);
      acc[ni] = wmma16(a, b, acc[ni]);
    }
    __syncthreads();
  }
  const int mOff = (lane >> 4) * 8;
  const int n0   = lane & 15;
#pragma unroll
  for (int ni = 0; ni < 4; ++ni)
#pragma unroll
    for (int r = 0; r < 8; ++r) {
      int row = mBase + wm * 16 + mOff + r;
      int col = nBase + wn * 64 + ni * 16 + n0;
      out[(size_t)row * C_DIM + col] = acc[ni][r] + bias[col];
    }
}

// ---------------- host side ----------------

extern "C" void kernel_launch(void* const* d_in, const int* in_sizes, int n_in,
                              void* d_out, int out_size, void* d_ws, size_t ws_size,
                              hipStream_t stream) {
  const float* xs1   = (const float*)d_in[0];
  const float* xs2   = (const float*)d_in[1];
  const float* xq    = (const float*)d_in[2];
  const float* Wqkv  = (const float*)d_in[3];
  const float* bqkv  = (const float*)d_in[4];
  const float* Wproj = (const float*)d_in[5];
  const float* bproj = (const float*)d_in[6];
  const float* rel   = (const float*)d_in[7];
  float* out = (float*)d_out;

  _Float16* qkv  = (_Float16*)d_ws;                                  // [49152,768] f16
  _Float16* attn = (_Float16*)((char*)d_ws +
                    (size_t)MTOT * QKV_N * sizeof(_Float16));        // [40960,256] f16

  qkv_gemm_kernel<<<dim3(MTOT / 64, QKV_N / 128), 256, 0, stream>>>(
      xs1, xs2, xq, Wqkv, bqkv, qkv);

  win_attn_kernel<<<512 * NHEAD, 32, 0, stream>>>(qkv, rel, attn);

  cross_attn_kernel<<<NBATCH * NHEAD * 16, 32, 0, stream>>>(qkv, attn);

  proj_gemm_kernel<<<dim3(OUT_MTOT / 64, C_DIM / 128), 256, 0, stream>>>(
      attn, Wproj, bproj, out);
}
